// FusedAttention_9397388444365
// MI455X (gfx1250) — matrix operand order repaired and checked by reading, not compiled, not run on hardware
//
#include <hip/hip_runtime.h>

namespace {
constexpr int B = 2, S = 2048, H = 16, D = 128;
constexpr int QTILE = 64, KTILE = 32, WAVES = 4;
constexpr float SCALE = 0.08838834764831845f;   // 1/sqrt(128)
constexpr float LOG2E = 1.44269504088896340736f;
}

typedef __attribute__((ext_vector_type(16))) __bf16 v16bf;
typedef __attribute__((ext_vector_type(8)))  __bf16 v8bf;
typedef __attribute__((ext_vector_type(8)))  float  v8f;

// native f32 -> bf16 convert (hardware v_cvt, not manual bit-twiddling)
static __device__ __forceinline__ __bf16 f2bf(float f) {
  return static_cast<__bf16>(f);
}
static __device__ __forceinline__ unsigned short f2bfu(float f) {
  return __builtin_bit_cast(unsigned short, static_cast<__bf16>(f));
}

__global__ __launch_bounds__(WAVES * 32)
void fa_fwd(const float* __restrict__ qg, const float* __restrict__ kg,
            const float* __restrict__ vg, const int* __restrict__ cuq,
            const int* __restrict__ cukv, float* __restrict__ outg)
{
  __shared__ __align__(128) unsigned short kLds[KTILE * D];        // [kv][d] bf16
  __shared__ __align__(128) unsigned short vLds[D * KTILE];        // [d][kv] bf16 (transposed)
  __shared__ __align__(128) unsigned short pLds[WAVES][16 * KTILE]; // per-wave P staging

  const int qb = blockIdx.x, h = blockIdx.y, b = blockIdx.z;
  const int tid  = threadIdx.x;
  const int wave = tid >> 5;
  const int lane = tid & 31;
  const int half = lane >> 4;   // K-half selector per 16-bit WMMA layouts
  const int lm   = lane & 15;

  const int len_q  = cuq[b + 1]  - cuq[b];
  const int len_kv = cukv[b + 1] - cukv[b];
  const int q0  = qb * QTILE;
  const int q0w = q0 + wave * 16;

  // ---- Q fragments: 16x128 per wave as 4 bf16 A-fragments (K chunks of 32)
  v16bf qf[4];
  {
    const int qrow = q0w + lm;
    const float* qptr = qg + (((size_t)b * S + qrow) * H + h) * D;
    #pragma unroll
    for (int c = 0; c < 4; ++c) {
      const float* p0 = qptr + c * 32 + half * 16;       // K = half*16 .. +7
      const float* p1 = qptr + c * 32 + half * 16 + 8;   // K = half*16+8 .. +7
      float4 f0 = *(const float4*)(p0);
      float4 f1 = *(const float4*)(p0 + 4);
      float4 f2 = *(const float4*)(p1);
      float4 f3 = *(const float4*)(p1 + 4);
      float t[16] = {f0.x,f0.y,f0.z,f0.w, f1.x,f1.y,f1.z,f1.w,
                     f2.x,f2.y,f2.z,f2.w, f3.x,f3.y,f3.z,f3.w};
      #pragma unroll
      for (int j = 0; j < 16; ++j)
        qf[c][j] = f2bf(t[j]);
    }
  }

  v8f acc[8];                 // O accumulator, 8 N-tiles of 16 (D=128)
  float m_i[8], l_i[8];       // per-row running max / sum (rows r + 8*half)
  #pragma unroll
  for (int r = 0; r < 8; ++r) { m_i[r] = -1e30f; l_i[r] = 0.0f; }
  #pragma unroll
  for (int n = 0; n < 8; ++n)
    #pragma unroll
    for (int r = 0; r < 8; ++r) acc[n][r] = 0.0f;

  const int qmax  = q0 + QTILE - 1;
  const int kvend = min(len_kv, qmax + 1);    // causal tile skip

  for (int k0 = 0; k0 < kvend; k0 += KTILE) {
    __syncthreads();
    // ---- cooperative K/V tile load, f32 -> bf16 (K row-major, V transposed)
    #pragma unroll
    for (int i = 0; i < 8; ++i) {
      const int lin = i * (WAVES * 32) + tid;   // float4 index, 1024 total
      const int row = lin >> 5;                 // 32 float4 per 128-wide row
      const int c4  = (lin & 31) * 4;
      const size_t g = (((size_t)b * S + k0 + row) * H + h) * D + c4;
      float4 kf = *(const float4*)(kg + g);
      float4 vf = *(const float4*)(vg + g);
      uint2 pk;
      pk.x = (unsigned)f2bfu(kf.x) | ((unsigned)f2bfu(kf.y) << 16);
      pk.y = (unsigned)f2bfu(kf.z) | ((unsigned)f2bfu(kf.w) << 16);
      *(uint2*)(&kLds[row * D + c4]) = pk;
      unsigned short vh[4] = {f2bfu(vf.x), f2bfu(vf.y), f2bfu(vf.z), f2bfu(vf.w)};
      #pragma unroll
      for (int j = 0; j < 4; ++j) vLds[(c4 + j) * KTILE + row] = vh[j];
    }
    if (k0 + KTILE < kvend) {   // prefetch next tile -> global_prefetch_b8
      const size_t g = (((size_t)b * S + k0 + KTILE + (tid >> 2)) * H + h) * D
                     + (size_t)(tid & 3) * 32;
      __builtin_prefetch(kg + g, 0, 1);
      __builtin_prefetch(vg + g, 0, 1);
    }
    __syncthreads();

    // ---- S = Q @ K^T : two 16x16 tiles over the 32 kv columns
    v8f s0, s1;
    #pragma unroll
    for (int r = 0; r < 8; ++r) { s0[r] = 0.0f; s1[r] = 0.0f; }
    #pragma unroll
    for (int c = 0; c < 4; ++c) {
      const v16bf bk0 = *(const v16bf*)(&kLds[lm * D        + c * 32 + half * 16]);
      const v16bf bk1 = *(const v16bf*)(&kLds[(16 + lm) * D + c * 32 + half * 16]);
      s0 = __builtin_amdgcn_wmma_f32_16x16x32_bf16(false, qf[c], false, bk0, (short)0, s0, false, false);
      s1 = __builtin_amdgcn_wmma_f32_16x16x32_bf16(false, qf[c], false, bk1, (short)0, s1, false, false);
    }

    // ---- masked online softmax (rows r + 8*half, cols lm / lm+16)
    unsigned short* pW = pLds[wave];
    #pragma unroll
    for (int r = 0; r < 8; ++r) {
      const int qrow = q0w + r + 8 * half;
      const int kv0 = k0 + lm, kv1 = k0 + 16 + lm;
      const bool vq  = qrow < len_q;
      const bool ok0 = vq && (kv0 <= qrow) && (kv0 < len_kv);
      const bool ok1 = vq && (kv1 <= qrow) && (kv1 < len_kv);
      const float a0 = ok0 ? s0[r] * SCALE : -1e30f;
      const float a1 = ok1 ? s1[r] * SCALE : -1e30f;
      float rmax = fmaxf(a0, a1);
      #pragma unroll
      for (int off = 1; off < 16; off <<= 1)
        rmax = fmaxf(rmax, __shfl_xor(rmax, off, 32));
      const float newm = fmaxf(m_i[r], rmax);
      const float p0 = ok0 ? __builtin_amdgcn_exp2f((a0 - newm) * LOG2E) : 0.0f;
      const float p1 = ok1 ? __builtin_amdgcn_exp2f((a1 - newm) * LOG2E) : 0.0f;
      const float cr = __builtin_amdgcn_exp2f((m_i[r] - newm) * LOG2E);
      float rsum = p0 + p1;
      #pragma unroll
      for (int off = 1; off < 16; off <<= 1)
        rsum += __shfl_xor(rsum, off, 32);
      l_i[r] = l_i[r] * cr + rsum;
      m_i[r] = newm;
      #pragma unroll
      for (int n = 0; n < 8; ++n) acc[n][r] *= cr;   // rescale running O
      pW[(r + 8 * half) * KTILE + lm]      = f2bfu(p0);
      pW[(r + 8 * half) * KTILE + 16 + lm] = f2bfu(p1);
    }

    // ---- P as A-fragment (C->A relayout through per-wave LDS), O += P @ V
    v16bf pf;
    {
      const v8bf pa = *(const v8bf*)(&pW[lm * KTILE + half * 16]);
      const v8bf pb = *(const v8bf*)(&pW[lm * KTILE + half * 16 + 8]);
      #pragma unroll
      for (int j = 0; j < 8; ++j) { pf[j] = pa[j]; pf[8 + j] = pb[j]; }
    }
    #pragma unroll
    for (int n = 0; n < 8; ++n) {
      const v16bf bv = *(const v16bf*)(&vLds[(n * 16 + lm) * KTILE + half * 16]);
      acc[n] = __builtin_amdgcn_wmma_f32_16x16x32_bf16(false, pf, false, bv, (short)0, acc[n], false, false);
    }
  }

  // ---- epilogue: normalize by l, zero-fill padded query rows, store O
  #pragma unroll
  for (int r = 0; r < 8; ++r) {
    const int qrow = q0w + r + 8 * half;
    const bool vq = (qrow < len_q) && (l_i[r] > 0.0f);
    const float inv = vq ? 1.0f / l_i[r] : 0.0f;
    float* ob = outg + (((size_t)b * S + qrow) * H + h) * D;
    #pragma unroll
    for (int n = 0; n < 8; ++n)
      ob[n * 16 + lm] = acc[n][r] * inv;
  }
}

extern "C" void kernel_launch(void* const* d_in, const int* in_sizes, int n_in,
                              void* d_out, int out_size, void* d_ws, size_t ws_size,
                              hipStream_t stream) {
  const float* q  = (const float*)d_in[0];
  const float* k  = (const float*)d_in[1];
  const float* v  = (const float*)d_in[2];
  const int* cuq  = (const int*)d_in[3];
  const int* cukv = (const int*)d_in[4];
  float* out = (float*)d_out;
  dim3 grid(S / QTILE, H, B);
  fa_fwd<<<grid, dim3(WAVES * 32), 0, stream>>>(q, k, v, cuq, cukv, out);
}
